// MultiHeadAttention_62843961475274
// MI455X (gfx1250) — matrix operand order repaired and checked by reading, not compiled, not run on hardware
//
#include <hip/hip_runtime.h>
#include <hip/hip_bf16.h>

// ---------------------------------------------------------------------------
// MHA block for gfx1250 (CDNA5, wave32, WMMA 16x16x32 f16 -> f32 accumulate)
// Fused scores+softmax+attnV keeps a 64x2048 f16 score stripe in LDS (320KB
// per WGP on CDNA5) so the 536MB attention matrix is written to HBM exactly
// once. Q tile staged into LDS via the Tensor Data Mover (TDM).
// ---------------------------------------------------------------------------

typedef _Float16 half_t;
typedef __attribute__((ext_vector_type(16))) _Float16 v16h;
typedef __attribute__((ext_vector_type(8)))  _Float16 v8h;
typedef __attribute__((ext_vector_type(8)))  float    v8f;
typedef unsigned int v4ui __attribute__((ext_vector_type(4)));
typedef int          v8i  __attribute__((ext_vector_type(8)));
typedef int          v4i  __attribute__((ext_vector_type(4)));

#define D_MODEL   1024
#define NUM_HEADS 16
#define D_K       64
#define SEQ       2048
#define BATCH     2
#define MROWS     (BATCH * SEQ)   // 4096

#define QTILE     64              // query rows per workgroup in fused kernel
#define LDSW      (SEQ + 8)       // f16 row stride: +16B pad spreads LDS banks
#define SC_BYTES  (QTILE * LDSW * 2)                      // 263168
#define QSTAGE_OFF (SC_BYTES + (QTILE + QTILE + 256) * 4) // 264704
#define QSTRIDE   72              // Q stage row stride (64 + 8 halves pad)
#define SMEM_BYTES (QSTAGE_OFF + QTILE * QSTRIDE * 2)     // 273920 <= 320KB

// ---------------------------------------------------------------------------
// WMMA fragment loaders (wave32; A/B use this code's contiguous k order, not the ISA table order)
// A 16x32 f16: lanes 0-15 -> row M=lane, K=0..15
//              lanes 16-31 -> row M=lane-16, K=16..31
// B 32x16 f16: lanes 0-15 -> col N=lane, K=0..15; lanes 16-31 -> K=16..31
// C/D 16x16 f32: vgpr v -> M = v + 8*(lane>=16), N = lane&15
// ---------------------------------------------------------------------------

static __device__ inline v16h load_a_frag_f32(const float* __restrict__ base,
                                              int stride, int row, int k0) {
    const int lane = threadIdx.x & 31;
    const int r  = lane & 15;
    const int kq = (lane >> 4) << 4;          // 0 or 16
    const float* p = base + (size_t)(row + r) * stride + k0 + kq;
    v16h f;
#pragma unroll
    for (int i = 0; i < 8; i += 4) {
        float4 x = *(const float4*)(p + i);
        f[i]     = (half_t)x.x; f[i + 1] = (half_t)x.y;
        f[i + 2] = (half_t)x.z; f[i + 3] = (half_t)x.w;
        float4 y = *(const float4*)(p + 8 + i);
        f[8 + i]  = (half_t)y.x; f[9 + i]  = (half_t)y.y;
        f[10 + i] = (half_t)y.z; f[11 + i] = (half_t)y.w;
    }
    return f;
}

static __device__ inline v16h load_a_frag_f16(const half_t* base,
                                              int stride, int row, int k0) {
    const int lane = threadIdx.x & 31;
    const int r  = lane & 15;
    const int kq = (lane >> 4) << 4;          // 0 or 16
    const half_t* p = base + (size_t)(row + r) * stride + k0 + kq;
    v8h lo = *(const v8h*)(p);                // K = kq .. kq+7
    v8h hi = *(const v8h*)(p + 8);            // K = kq+8 .. kq+15
    return __builtin_shufflevector(lo, hi, 0, 1, 2, 3, 4, 5, 6, 7,
                                           8, 9, 10, 11, 12, 13, 14, 15);
}

static __device__ inline v16h load_b_frag_f16(const half_t* __restrict__ base,
                                              int stride, int col, int k0) {
    const int lane = threadIdx.x & 31;
    const int c  = lane & 15;
    const int kh = (lane >> 4) << 4;          // 0 or 16
    const half_t* p = base + (size_t)(col + c) * stride + k0 + kh;
    return *(const v16h*)p;                   // 16 contiguous K values
}

static __device__ inline v16h load_b_frag_f32(const float* __restrict__ base,
                                              int stride, int col, int k0) {
    const int lane = threadIdx.x & 31;
    const int c  = lane & 15;
    const int kh = (lane >> 4) << 4;          // 0 or 16
    const float* p = base + (size_t)(col + c) * stride + k0 + kh;
    v16h f;
#pragma unroll
    for (int i = 0; i < 16; i += 4) {
        float4 x = *(const float4*)(p + i);
        f[i]     = (half_t)x.x; f[i + 1] = (half_t)x.y;
        f[i + 2] = (half_t)x.z; f[i + 3] = (half_t)x.w;
    }
    return f;
}

static __device__ inline v8f wmma_f16(v16h a, v16h b, v8f c) {
    return __builtin_amdgcn_wmma_f32_16x16x32_f16(false, a, false, b,
                                                  (short)0, c, false, false);
}

static __device__ inline v8f zero_v8f() {
    v8f z = {0.f, 0.f, 0.f, 0.f, 0.f, 0.f, 0.f, 0.f};
    return z;
}

// ---------------------------------------------------------------------------
// Kernel 1: projection  Out = X @ W^T + bias, f16 output.
// TRANS_V stores into [B, H, Dk, S] so attn@V sees a K-contiguous B operand.
// ---------------------------------------------------------------------------
template <bool TRANS_V>
__global__ __launch_bounds__(256) void proj_kernel(const float* __restrict__ X,
                                                   const float* __restrict__ W,
                                                   const float* __restrict__ bias,
                                                   half_t* __restrict__ Out) {
    const int wave = threadIdx.x >> 5;
    const int lane = threadIdx.x & 31;
    const int row  = blockIdx.y * 128 + wave * 16;
    const int col  = blockIdx.x * 64;

    v8f acc[4] = {zero_v8f(), zero_v8f(), zero_v8f(), zero_v8f()};

    for (int k0 = 0; k0 < D_MODEL; k0 += 32) {
        v16h a = load_a_frag_f32(X, D_MODEL, row, k0);
#pragma unroll
        for (int t = 0; t < 4; ++t) {
            v16h b = load_b_frag_f32(W, D_MODEL, col + 16 * t, k0);
            acc[t] = wmma_f16(a, b, acc[t]);
        }
    }

    const int n  = lane & 15;
    const int m0 = (lane >> 4) << 3;
#pragma unroll
    for (int t = 0; t < 4; ++t) {
        const int   cg = col + 16 * t + n;
        const float bv = bias[cg];
#pragma unroll
        for (int v = 0; v < 8; ++v) {
            const float val = acc[t][v] + bv;
            const int   mg  = row + m0 + v;
            if constexpr (TRANS_V) {
                const int b  = mg >> 11;          // / SEQ
                const int s  = mg & (SEQ - 1);
                const int hh = cg >> 6;           // / D_K
                const int dk = cg & (D_K - 1);
                Out[((size_t)(b * NUM_HEADS + hh) * D_K + dk) * SEQ + s] =
                    (half_t)val;
            } else {
                Out[(size_t)mg * D_MODEL + cg] = (half_t)val;
            }
        }
    }
}

// ---------------------------------------------------------------------------
// Kernel 2: fused attention per (b,h, 64-query tile).
//   phase 0: stage Q tile (64x64 f16) into LDS via TDM tensor_load_to_lds
//            (D# per CDNA5 ISA; TDM pad feature
//            inserts 4 dwords per 32-dword row -> stride 72 halves, banks
//            spread). Fallback: manual vectorized copy.
//   phase 1: scores = (Q Kh^T)/8 via WMMA -> f16 stripe in LDS (64 x 2048)
//   phase 2: row max / exp / row sum, all v8h-vectorized LDS traffic;
//            probs -> d_out exactly once, b128-coalesced; LDS keeps
//            UNnormalized exp (scale folded into phase 3 store)
//   phase 3: O = attn @ V via WMMA, A-fragments re-read from LDS.
// ---------------------------------------------------------------------------
__global__ __launch_bounds__(256) void fused_attn_kernel(
        const half_t* __restrict__ Qh, const half_t* __restrict__ Kh,
        const half_t* __restrict__ Vt, float* __restrict__ attn,
        half_t* __restrict__ Oh) {
    extern __shared__ char smem_raw[];
    half_t* sc     = (half_t*)smem_raw;                     // [QTILE][LDSW]
    float*  rmax   = (float*)(smem_raw + SC_BYTES);         // [QTILE]
    float*  rinv   = rmax + QTILE;                          // [QTILE]
    float*  pred   = rinv + QTILE;                          // [256]
    half_t* qstage = (half_t*)(smem_raw + QSTAGE_OFF);      // [QTILE][QSTRIDE]

    const int bh = blockIdx.y;
    const int b  = bh >> 4;                  // / NUM_HEADS
    const int h  = bh & (NUM_HEADS - 1);
    const int qbase = blockIdx.x * QTILE;

    const half_t* Qb = Qh + (size_t)b * SEQ * D_MODEL + h * D_K;
    const half_t* Kb = Kh + (size_t)b * SEQ * D_MODEL + h * D_K;
    const half_t* Bv = Vt + (size_t)bh * D_K * SEQ;
    float* outp = attn + (size_t)bh * SEQ * SEQ;

    const int tid  = threadIdx.x;
    const int wave = tid >> 5;
    const int lane = tid & 31;
    const int n    = lane & 15;
    const int m0   = (lane >> 4) << 3;

    // ---- phase 0: stage Q tile into LDS
#if defined(__gfx1250__) && __has_builtin(__builtin_amdgcn_tensor_load_to_lds) \
                         && __has_builtin(__builtin_amdgcn_s_wait_tensorcnt)
    if (wave == 0) {
        const unsigned long long ga =
            (unsigned long long)(uintptr_t)(Qb + (size_t)qbase * D_MODEL);
        v4ui g0;
        g0[0] = 1u;                                   // count=1, user desc
        g0[1] = (unsigned)QSTAGE_OFF;                 // lds_addr (bytes)
        g0[2] = (unsigned)(ga & 0xFFFFFFFFu);         // global_addr lo
        g0[3] = (unsigned)((ga >> 32) & 0x1FFFFFFu)   // global_addr hi (56:32)
              | (2u << 30);                           // type = 2 ("image")
        v8i g1;
        g1[0] = (1 << 16)    // data_size = 1 -> 2 bytes
              | (1 << 20)    // pad_enable
              | (4 << 22)    // pad_interval: 2^(4+1) = 32 dwords (one row)
              | (3 << 25);   // pad_amount: 4 dwords -> row stride 72 halves
        g1[1] = (64 << 16);  // tensor_dim0 = 64 (bits 79:48, low half)
        g1[2] = (64 << 16);  // tensor_dim1 = 64 (bits 111:80, low half)
        g1[3] = (64 << 16);  // tile_dim0   = 64 (bits 127:112)
        g1[4] = 64;          // tile_dim1   = 64 (bits 143:128); tile_dim2 = 0
        g1[5] = D_MODEL;     // tensor_dim0_stride = 1024 (bits 207:160, lo32)
        g1[6] = 0;           // dim0_stride hi / dim1_stride lo
        g1[7] = 0;
        v4i gz  = {0, 0, 0, 0};
        v8i gz8 = {0, 0, 0, 0, 0, 0, 0, 0};
        // 6-arg form on this toolchain: (g0, g1, g2, g3, g4, cpol)
        __builtin_amdgcn_tensor_load_to_lds(g0, g1, gz, gz, gz8, 0);
        __builtin_amdgcn_s_wait_tensorcnt(0);
    }
#else
    for (int idx = tid; idx < QTILE * 8; idx += 256) {
        const int r   = idx >> 3;
        const int ch8 = (idx & 7) * 8;
        *(v8h*)(qstage + r * QSTRIDE + ch8) =
            *(const v8h*)(Qb + (size_t)(qbase + r) * D_MODEL + ch8);
    }
#endif
    __syncthreads();

    // ---- preload Q fragments from LDS (K = 64 -> 2 k-steps)
    v16h aq[4][2];
#pragma unroll
    for (int mi = 0; mi < 4; ++mi)
#pragma unroll
        for (int ks = 0; ks < 2; ++ks)
            aq[mi][ks] = load_a_frag_f16(qstage, QSTRIDE, 16 * mi, 32 * ks);

    // ---- phase 1: waves stripe over the 128 key tiles
    for (int nt = wave; nt < SEQ / 16; nt += 8) {
        const int col = nt * 16;
        if (nt + 8 < SEQ / 16) {  // prefetch next K tile (global_prefetch_b8)
            __builtin_prefetch(Kb + (size_t)(col + 128 + n) * D_MODEL +
                               ((lane >> 4) << 4), 0, 1);
        }
        v16h bk0 = load_b_frag_f16(Kb, D_MODEL, col, 0);
        v16h bk1 = load_b_frag_f16(Kb, D_MODEL, col, 32);
#pragma unroll
        for (int mi = 0; mi < 4; ++mi) {
            v8f acc = zero_v8f();
            acc = wmma_f16(aq[mi][0], bk0, acc);
            acc = wmma_f16(aq[mi][1], bk1, acc);
#pragma unroll
            for (int v = 0; v < 8; ++v)
                sc[(mi * 16 + m0 + v) * LDSW + col + n] =
                    (half_t)(acc[v] * 0.125f);
        }
    }
    __syncthreads();

    // ---- phase 2a: row max (4 threads/row, 512 cols each, v8h vectorized)
    const int r2   = tid >> 2;
    const int seg  = tid & 3;
    half_t* rowp2  = sc + r2 * LDSW + seg * (SEQ / 4);
    float mx = -3.402823466e38f;
    for (int c = 0; c < SEQ / 4; c += 8) {
        v8h hv = *(const v8h*)(rowp2 + c);
#pragma unroll
        for (int i = 0; i < 8; ++i) mx = fmaxf(mx, (float)hv[i]);
    }
    pred[tid] = mx;
    __syncthreads();
    if (seg == 0)
        rmax[r2] = fmaxf(fmaxf(pred[tid], pred[tid + 1]),
                         fmaxf(pred[tid + 2], pred[tid + 3]));
    __syncthreads();

    // ---- phase 2b: exp + row sum; LDS keeps unnormalized exp (f16)
    const float rm = rmax[r2];
    float s = 0.f;
    for (int c = 0; c < SEQ / 4; c += 8) {
        v8h hv = *(const v8h*)(rowp2 + c);
        v8h ev;
#pragma unroll
        for (int i = 0; i < 8; ++i) {
            const float e = __expf((float)hv[i] - rm);
            ev[i] = (half_t)e;
            s += e;
        }
        *(v8h*)(rowp2 + c) = ev;
    }
    pred[tid] = s;
    __syncthreads();
    if (seg == 0)
        rinv[r2] = 1.0f / (pred[tid] + pred[tid + 1] +
                           pred[tid + 2] + pred[tid + 3]);
    __syncthreads();

    // ---- phase 2c: probs -> d_out once; thread t owns 8 contiguous cols
    //      (1 ds_load_b128 + 2 coalesced global b128 stores per row)
    for (int r = 0; r < QTILE; ++r) {
        const float inv = rinv[r];
        const v8h hv = *(const v8h*)(sc + r * LDSW + tid * 8);
        float4 f0, f1;
        f0.x = (float)hv[0] * inv; f0.y = (float)hv[1] * inv;
        f0.z = (float)hv[2] * inv; f0.w = (float)hv[3] * inv;
        f1.x = (float)hv[4] * inv; f1.y = (float)hv[5] * inv;
        f1.z = (float)hv[6] * inv; f1.w = (float)hv[7] * inv;
        float* orow = outp + (size_t)(qbase + r) * SEQ + tid * 8;
        *(float4*)(orow)     = f0;
        *(float4*)(orow + 4) = f1;
    }

    // ---- phase 3: O = exp-stripe @ V, A from LDS, scale by 1/rowsum
    const int mi = wave & 3;                 // M sub-tile (16 rows)
    const int ch = (wave >> 2) * 32;         // N half (32 of 64 cols)
    v8f o0 = zero_v8f(), o1 = zero_v8f();
    for (int k0 = 0; k0 < SEQ; k0 += 32) {
        v16h a  = load_a_frag_f16(sc, LDSW, mi * 16, k0);      // ds_load
        v16h b0 = load_b_frag_f16(Bv, SEQ, ch, k0);
        v16h b1 = load_b_frag_f16(Bv, SEQ, ch + 16, k0);
        o0 = wmma_f16(a, b0, o0);
        o1 = wmma_f16(a, b1, o1);
    }
    half_t* Ob = Oh + (size_t)b * SEQ * D_MODEL + h * D_K;
#pragma unroll
    for (int v = 0; v < 8; ++v) {
        const int   rl  = mi * 16 + m0 + v;
        const float inv = rinv[rl];
        half_t* prow = Ob + (size_t)(qbase + rl) * D_MODEL;
        prow[ch + n]      = (half_t)(o0[v] * inv);
        prow[ch + 16 + n] = (half_t)(o1[v] * inv);
    }
}

// ---------------------------------------------------------------------------
// Kernel 3: output projection  Xp = Oh @ Wo^T + bo  (fp32 out to scratch)
// ---------------------------------------------------------------------------
__global__ __launch_bounds__(256) void outproj_kernel(const half_t* __restrict__ Oh,
                                                      const float* __restrict__ Wo,
                                                      const float* __restrict__ bo,
                                                      float* __restrict__ Xp) {
    const int wave = threadIdx.x >> 5;
    const int lane = threadIdx.x & 31;
    const int row  = blockIdx.y * 128 + wave * 16;
    const int col  = blockIdx.x * 64;

    v8f acc[4] = {zero_v8f(), zero_v8f(), zero_v8f(), zero_v8f()};

    for (int k0 = 0; k0 < D_MODEL; k0 += 32) {
        v16h a = load_a_frag_f16(Oh, D_MODEL, row, k0);
#pragma unroll
        for (int t = 0; t < 4; ++t) {
            v16h b = load_b_frag_f32(Wo, D_MODEL, col + 16 * t, k0);
            acc[t] = wmma_f16(a, b, acc[t]);
        }
    }

    const int n  = lane & 15;
    const int m0 = (lane >> 4) << 3;
#pragma unroll
    for (int t = 0; t < 4; ++t) {
        const int   cg = col + 16 * t + n;
        const float bv = bo[cg];
#pragma unroll
        for (int v = 0; v < 8; ++v)
            Xp[(size_t)(row + m0 + v) * D_MODEL + cg] = acc[t][v] + bv;
    }
}

// ---------------------------------------------------------------------------
// Kernel 4: residual + LayerNorm. One block per row of 1024.
// ---------------------------------------------------------------------------
__global__ __launch_bounds__(256) void ln_kernel(const float* __restrict__ Xp,
                                                 const float* __restrict__ qin,
                                                 const float* __restrict__ g,
                                                 const float* __restrict__ beta,
                                                 float* __restrict__ out) {
    const size_t rb = (size_t)blockIdx.x * D_MODEL;
    __shared__ float red[256];
    const int t = threadIdx.x;

    float x[4];
    float s = 0.f;
#pragma unroll
    for (int i = 0; i < 4; ++i) {
        const int c = t + 256 * i;
        x[i] = Xp[rb + c] + qin[rb + c];
        s += x[i];
    }
    red[t] = s;
    __syncthreads();
    for (int off = 128; off > 0; off >>= 1) {
        if (t < off) red[t] += red[t + off];
        __syncthreads();
    }
    const float mu = red[0] * (1.0f / D_MODEL);
    __syncthreads();

    float vs = 0.f;
#pragma unroll
    for (int i = 0; i < 4; ++i) {
        const float d = x[i] - mu;
        vs += d * d;
    }
    red[t] = vs;
    __syncthreads();
    for (int off = 128; off > 0; off >>= 1) {
        if (t < off) red[t] += red[t + off];
        __syncthreads();
    }
    const float inv = rsqrtf(red[0] * (1.0f / D_MODEL) + 1e-5f);
#pragma unroll
    for (int i = 0; i < 4; ++i) {
        const int c = t + 256 * i;
        out[rb + c] = (x[i] - mu) * inv * g[c] + beta[c];
    }
}

// ---------------------------------------------------------------------------
// Launch
// ---------------------------------------------------------------------------
extern "C" void kernel_launch(void* const* d_in, const int* in_sizes, int n_in,
                              void* d_out, int out_size, void* d_ws, size_t ws_size,
                              hipStream_t stream) {
    const float* q    = (const float*)d_in[0];
    const float* k    = (const float*)d_in[1];
    const float* v    = (const float*)d_in[2];
    const float* Wq   = (const float*)d_in[3];
    const float* bq   = (const float*)d_in[4];
    const float* Wk   = (const float*)d_in[5];
    const float* bk   = (const float*)d_in[6];
    const float* Wv   = (const float*)d_in[7];
    const float* bv   = (const float*)d_in[8];
    const float* Wo   = (const float*)d_in[9];
    const float* bo   = (const float*)d_in[10];
    const float* ln_g = (const float*)d_in[11];
    const float* ln_b = (const float*)d_in[12];

    float* out_x    = (float*)d_out;
    float* out_attn = out_x + (size_t)BATCH * SEQ * D_MODEL;   // [B,H,S,S]

    const size_t NE = (size_t)MROWS * D_MODEL;   // 4096*1024
    half_t* Qh = (half_t*)d_ws;     // f16 Q  [B,S,D]            8 MB
    half_t* Kh = Qh + NE;           // f16 K  [B,S,D]            8 MB
    half_t* Vt = Kh + NE;           // f16 V  [B,H,Dk,S]         8 MB
    half_t* Oh = Vt + NE;           // f16 attn@V [B,S,D]        8 MB
    float*  Xp = (float*)(Oh + NE); // fp32 out-proj [B,S,D]    16 MB

    // opt in to >64KB dynamic LDS for the fused kernel (CDNA5 WGP: 320KB)
    (void)hipFuncSetAttribute((const void*)fused_attn_kernel,
                              hipFuncAttributeMaxDynamicSharedMemorySize,
                              SMEM_BYTES);

    dim3 blk(256);
    dim3 gproj(D_MODEL / 64, MROWS / 128);               // (16, 32)
    proj_kernel<false><<<gproj, blk, 0, stream>>>(q, Wq, bq, Qh);
    proj_kernel<false><<<gproj, blk, 0, stream>>>(k, Wk, bk, Kh);
    proj_kernel<true ><<<gproj, blk, 0, stream>>>(v, Wv, bv, Vt);

    dim3 gf(SEQ / QTILE, BATCH * NUM_HEADS);             // (32, 32)
    fused_attn_kernel<<<gf, blk, SMEM_BYTES, stream>>>(Qh, Kh, Vt,
                                                       out_attn, Oh);

    outproj_kernel<<<gproj, blk, 0, stream>>>(Oh, Wo, bo, Xp);

    ln_kernel<<<dim3(MROWS), blk, 0, stream>>>(Xp, q, ln_g, ln_b, out_x);
}
